// MapSegmentationHeadSRTransformer_45990509805894
// MI455X (gfx1250) — matrix operand order repaired and checked by reading, not compiled, not run on hardware
//
#include <hip/hip_runtime.h>
#include <hip/hip_bf16.h>

// ---------------------------------------------------------------------------
// MI455X (gfx1250). Compute-bound pipeline (~435 GFLOP) -> all GEMM/conv through
// v_wmma_f32_16x16x32_f16. Block tile 64x128x32, double-buffered LDS staged with
// CDNA5 async global->LDS copies (ASYNCcnt, partial waits for pipelining),
// 8 waves x (2x2) 16x16 WMMA tiles per block. Staging pointers advance
// incrementally (no per-iteration muls / exec-mask selects in the hot loop).
// A and B fragments both use contiguous k order: k = 16*(lane/16) + s.
// ---------------------------------------------------------------------------

typedef __attribute__((ext_vector_type(16))) _Float16 v16h;
typedef __attribute__((ext_vector_type(8)))  float    v8f;

#define LDS_STRIDE 36   // floats per row (32 data + 4 pad), keeps 16B alignment
#define A_BUF_BYTES (64 * LDS_STRIDE * 4)
#define B_BUF_BYTES (128 * LDS_STRIDE * 4)

__device__ __forceinline__ v8f wmma16x16x32(v16h a, v16h b, v8f c) {
  return __builtin_amdgcn_wmma_f32_16x16x32_f16(
      false, a, false, b, (short)0, c, false, false);
}

__device__ __forceinline__ void async_ld_b32(unsigned lds_off, const float* gp) {
  asm volatile("global_load_async_to_lds_b32 %0, %1, off"
               :: "v"(lds_off), "v"(gp) : "memory");
}
__device__ __forceinline__ void async_ld_b128(unsigned lds_off, const float* gp) {
  asm volatile("global_load_async_to_lds_b128 %0, %1, off"
               :: "v"(lds_off), "v"(gp) : "memory");
}
__device__ __forceinline__ void wait_async0() {
  asm volatile("s_wait_asynccnt 0x0" ::: "memory");
}
// Partial waits: async loads retire in order per wave, so <=batch outstanding
// means the oldest staged batch has fully landed while the newest keeps flying.
__device__ __forceinline__ void wait_async_le24() {
  asm volatile("s_wait_asynccnt 0x18" ::: "memory");
}
__device__ __forceinline__ void wait_async_le12() {
  asm volatile("s_wait_asynccnt 0xc" ::: "memory");
}

__device__ __forceinline__ v16h frag_a(const float* row, int lane) {
  const int kb = (lane & 16) ? 16 : 0;
  const float* p = row + kb;
  v16h a;
#pragma unroll
  for (int h = 0; h < 8; ++h) a[h] = (_Float16)p[h];
#pragma unroll
  for (int h = 0; h < 8; ++h) a[8 + h] = (_Float16)p[8 + h];
  return a;
}
__device__ __forceinline__ v16h frag_b(const float* row, int lane) {
  const int ks = (lane & 16) ? 16 : 0;
  const float* p = row + ks;
  v16h b;
#pragma unroll
  for (int h = 0; h < 16; ++h) b[h] = (_Float16)p[h];
  return b;
}

// ---------------------------------------------------------------------------
// Generic strided batched GEMM: Out(m,n)=act(alpha*sum_k A(m,k)B(k,n)+bias+Res)
// act: 0=none 1=relu 2=sigmoid.  256 thr, tile 64x128, K step 32, 2-deep pipe.
// ---------------------------------------------------------------------------
__global__ __launch_bounds__(256) void gemm_wmma_kernel(
    const float* __restrict__ A, long lam, long lak, long batA,
    const float* __restrict__ B, long lbk, long lbn, long batB,
    float* __restrict__ Out, long lom, long lon, long batO,
    const float* __restrict__ bias,
    const float* __restrict__ Res, long lrm, long lrn, long batR,
    const float* __restrict__ Zp,
    int M, int N, int K, float alpha, int act)
{
  __shared__ __align__(16) float As[2][64 * LDS_STRIDE];
  __shared__ __align__(16) float Bs[2][128 * LDS_STRIDE];
  const int tid = threadIdx.x;
  const int lane = tid & 31;
  const int wave = tid >> 5;
  const int wm = wave >> 2;
  const int wn = wave & 3;
  const long z = blockIdx.z;
  A += z * batA; B += z * batB; Out += z * batO;
  if (Res) Res += z * batR;
  const int m0 = blockIdx.x * 64;
  const int n0 = blockIdx.y * 128;

  // Per-thread staging: A 2048 elems (8/thr), B 4096 (16/thr).
  // Running pointers + per-slot steps (0 for OOB slots pinned to zero page).
  const float* aptr[8]; long astep[8]; unsigned aoff[8];
#pragma unroll
  for (int j = 0; j < 8; ++j) {
    const int e = wave * 256 + j * 32 + lane;
    const int m = e >> 5, k = e & 31;
    aoff[j] = (unsigned)(uintptr_t)&As[0][m * LDS_STRIDE + k];
    const bool ok = (m0 + m) < M;
    aptr[j] = ok ? (A + (long)(m0 + m) * lam + (long)k * lak) : Zp;
    astep[j] = ok ? 32 * lak : 0;
  }
  const float* bptr[16]; long bstep[16]; unsigned boff[16];
#pragma unroll
  for (int j = 0; j < 16; ++j) {
    const int e = wave * 512 + j * 32 + lane;
    const int n = e >> 5, k = e & 31;
    boff[j] = (unsigned)(uintptr_t)&Bs[0][n * LDS_STRIDE + k];
    const bool ok = (n0 + n) < N;
    bptr[j] = ok ? (B + (long)k * lbk + (long)(n0 + n) * lbn) : Zp;
    bstep[j] = ok ? 32 * lbk : 0;
  }

  auto stage = [&](int bufi) {
    const unsigned ao = bufi ? (unsigned)A_BUF_BYTES : 0u;
    const unsigned bo = bufi ? (unsigned)B_BUF_BYTES : 0u;
#pragma unroll
    for (int j = 0; j < 8; ++j) {
      async_ld_b32(aoff[j] + ao, aptr[j]);
      aptr[j] += astep[j];
    }
#pragma unroll
    for (int j = 0; j < 16; ++j) {
      async_ld_b32(boff[j] + bo, bptr[j]);
      bptr[j] += bstep[j];
    }
  };

  v8f acc[2][2] = {};
  stage(0);
  int buf = 0;
  for (int k0 = 0; k0 < K; k0 += 32) {
    if (k0 + 32 < K) {
      stage(buf ^ 1);        // keep next batch in flight behind the WMMAs
      wait_async_le24();     // oldest batch (current buf) has landed
    } else {
      wait_async0();
    }
    __syncthreads();

    const float* ab = &As[buf][0];
    const float* bb = &Bs[buf][0];
    const v16h a0 = frag_a(ab + (wm * 32 + (lane & 15)) * LDS_STRIDE, lane);
    const v16h a1 = frag_a(ab + (wm * 32 + 16 + (lane & 15)) * LDS_STRIDE, lane);
    const v16h b0 = frag_b(bb + (wn * 32 + (lane & 15)) * LDS_STRIDE, lane);
    const v16h b1 = frag_b(bb + (wn * 32 + 16 + (lane & 15)) * LDS_STRIDE, lane);
    acc[0][0] = wmma16x16x32(a0, b0, acc[0][0]);
    acc[0][1] = wmma16x16x32(a0, b1, acc[0][1]);
    acc[1][0] = wmma16x16x32(a1, b0, acc[1][0]);
    acc[1][1] = wmma16x16x32(a1, b1, acc[1][1]);
    __syncthreads();         // readers done before buf is re-staged
    buf ^= 1;
  }

#pragma unroll
  for (int i = 0; i < 2; ++i)
#pragma unroll
    for (int j = 0; j < 2; ++j) {
      const int n = n0 + wn * 32 + j * 16 + (lane & 15);
      if (n >= N) continue;
      const float bv = bias ? bias[n] : 0.f;
#pragma unroll
      for (int v = 0; v < 8; ++v) {
        const int m = m0 + wm * 32 + i * 16 + v + ((lane & 16) ? 8 : 0);
        if (m >= M) continue;
        float val = acc[i][j][v] * alpha + bv;
        if (Res) val += Res[(long)m * lrm + (long)n * lrn];
        if (act == 1) val = fmaxf(val, 0.f);
        else if (act == 2) val = 1.f / (1.f + __expf(-val));
        Out[(long)m * lom + (long)n * lon] = val;
      }
    }
}

// ---------------------------------------------------------------------------
// Implicit-GEMM conv3x3 (pad=1): X (Cin,H,W) * Wt (Cout,Cin,3,3) -> Out (Cout,H,W)
// M=H*W, N=Cout, K=Cin*9. Same 64x128x32 double-buffered tiling; halo/OOB via
// branchless clamped-address + zero-page select. Incremental (ic,rs) update:
// k += 32  <=>  ic += 3, rs += 5 (mod 9). Weights staged with async b128.
// ---------------------------------------------------------------------------
__global__ __launch_bounds__(256) void conv3x3_wmma_kernel(
    const float* __restrict__ X, const float* __restrict__ Wt,
    float* __restrict__ Out,
    const float* __restrict__ bias,
    const float* __restrict__ Res,
    const float* __restrict__ bng, const float* __restrict__ bnb,
    const float* __restrict__ bnm, const float* __restrict__ bnv,
    const float* __restrict__ Zp,
    int Cin, int H, int W, int Cout, int relu)
{
  const int M = H * W;
  const int K = Cin * 9;
  __shared__ __align__(16) float As[2][64 * LDS_STRIDE];
  __shared__ __align__(16) float Bs[2][128 * LDS_STRIDE];
  const int tid = threadIdx.x;
  const int lane = tid & 31;
  const int wave = tid >> 5;
  const int wm = wave >> 2;
  const int wn = wave & 3;
  const int m0 = blockIdx.x * 64;
  const int n0 = blockIdx.y * 128;

  unsigned aoff[8]; int aoh[8], aow[8], aic[8], ars[8]; bool aok[8];
#pragma unroll
  for (int j = 0; j < 8; ++j) {
    const int e = wave * 256 + j * 32 + lane;
    const int m = e >> 5, k = e & 31;
    aoff[j] = (unsigned)(uintptr_t)&As[0][m * LDS_STRIDE + k];
    aic[j] = k / 9;
    ars[j] = k - aic[j] * 9;
    const int gm = m0 + m;
    aok[j] = gm < M;
    const int mm = aok[j] ? gm : 0;
    aoh[j] = mm / W;
    aow[j] = mm - (mm / W) * W;
  }
  const float* bptr[4]; long bstep[4]; unsigned boff[4];
#pragma unroll
  for (int j = 0; j < 4; ++j) {
    const int g = wave * 128 + j * 32 + lane;
    const int n = g >> 3, kq = (g & 7) * 4;
    boff[j] = (unsigned)(uintptr_t)&Bs[0][n * LDS_STRIDE + kq];
    const bool ok = (n0 + n) < Cout;
    bptr[j] = ok ? (Wt + (long)(n0 + n) * K + kq) : Zp;
    bstep[j] = ok ? 32 : 0;
  }

  auto stage = [&](int bufi) {
    const unsigned ao = bufi ? (unsigned)A_BUF_BYTES : 0u;
    const unsigned bo = bufi ? (unsigned)B_BUF_BYTES : 0u;
#pragma unroll
    for (int j = 0; j < 8; ++j) {
      const int r = ars[j] / 3;
      const int s = ars[j] - r * 3;
      const int ih = aoh[j] + r - 1;
      const int iw = aow[j] + s - 1;
      const bool ok = aok[j] && ih >= 0 && ih < H && iw >= 0 && iw < W;
      const int ihc = ih < 0 ? 0 : (ih >= H ? H - 1 : ih);
      const int iwc = iw < 0 ? 0 : (iw >= W ? W - 1 : iw);
      const float* pin = X + (long)aic[j] * M + (long)ihc * W + iwc;
      async_ld_b32(aoff[j] + ao, ok ? pin : Zp);
      aic[j] += 3;                                  // advance k by 32 = 3*9+5
      ars[j] += 5;
      if (ars[j] >= 9) { ars[j] -= 9; aic[j] += 1; }
    }
#pragma unroll
    for (int j = 0; j < 4; ++j) {
      async_ld_b128(boff[j] + bo, bptr[j]);
      bptr[j] += bstep[j];
    }
  };

  v8f acc[2][2] = {};
  stage(0);
  int buf = 0;
  for (int k0 = 0; k0 < K; k0 += 32) {
    if (k0 + 32 < K) {
      stage(buf ^ 1);
      wait_async_le12();
    } else {
      wait_async0();
    }
    __syncthreads();

    const float* ab = &As[buf][0];
    const float* bb = &Bs[buf][0];
    const v16h a0 = frag_a(ab + (wm * 32 + (lane & 15)) * LDS_STRIDE, lane);
    const v16h a1 = frag_a(ab + (wm * 32 + 16 + (lane & 15)) * LDS_STRIDE, lane);
    const v16h b0 = frag_b(bb + (wn * 32 + (lane & 15)) * LDS_STRIDE, lane);
    const v16h b1 = frag_b(bb + (wn * 32 + 16 + (lane & 15)) * LDS_STRIDE, lane);
    acc[0][0] = wmma16x16x32(a0, b0, acc[0][0]);
    acc[0][1] = wmma16x16x32(a0, b1, acc[0][1]);
    acc[1][0] = wmma16x16x32(a1, b0, acc[1][0]);
    acc[1][1] = wmma16x16x32(a1, b1, acc[1][1]);
    __syncthreads();
    buf ^= 1;
  }

#pragma unroll
  for (int i = 0; i < 2; ++i)
#pragma unroll
    for (int j = 0; j < 2; ++j) {
      const int n = n0 + wn * 32 + j * 16 + (lane & 15);
      if (n >= Cout) continue;
      const float bv = bias ? bias[n] : 0.f;
      float sc = 1.f, sh = 0.f, mu = 0.f;
      if (bng) { sc = bng[n] * rsqrtf(bnv[n] + 1e-5f); sh = bnb[n]; mu = bnm[n]; }
#pragma unroll
      for (int v = 0; v < 8; ++v) {
        const int m = m0 + wm * 32 + i * 16 + v + ((lane & 16) ? 8 : 0);
        if (m >= M) continue;
        float val = acc[i][j][v] + bv;
        if (bng) val = (val - mu) * sc + sh;
        if (Res) val += Res[(long)n * M + m];
        if (relu) val = fmaxf(val, 0.f);
        Out[(long)n * M + m] = val;
      }
    }
}

// ---------------------------------------------------------------------------
__global__ __launch_bounds__(256) void softmax_rows_kernel(float* __restrict__ S, int n) {
  float* p = S + (long)blockIdx.x * n;
  __shared__ float sm[256];
  const int tid = threadIdx.x;
  float mx = -3.402823466e+38f;
  for (int i = tid; i < n; i += 256) mx = fmaxf(mx, p[i]);
  sm[tid] = mx; __syncthreads();
  for (int s = 128; s > 0; s >>= 1) { if (tid < s) sm[tid] = fmaxf(sm[tid], sm[tid + s]); __syncthreads(); }
  mx = sm[0]; __syncthreads();
  float sum = 0.f;
  for (int i = tid; i < n; i += 256) { float e = __expf(p[i] - mx); p[i] = e; sum += e; }
  sm[tid] = sum; __syncthreads();
  for (int s = 128; s > 0; s >>= 1) { if (tid < s) sm[tid] += sm[tid + s]; __syncthreads(); }
  const float inv = 1.f / sm[0];
  for (int i = tid; i < n; i += 256) p[i] *= inv;
}

__global__ __launch_bounds__(256) void grid_sample_kernel(
    const float* __restrict__ X, float* __restrict__ Out,
    int Hin, int Win, int Hout, int Wout, int C)
{
  const int idx = blockIdx.x * blockDim.x + threadIdx.x;
  const int total = C * Hout * Wout;
  if (idx >= total) return;
  const int ox = idx % Wout;
  const int t  = idx / Wout;
  const int oy = t % Hout;
  const int c  = t / Hout;
  const float vy = -49.f + 2.f * (float)oy;
  const float gy = (vy + 51.2f) * (1.f / 102.4f) * 2.f - 1.f;
  const float fy = ((gy + 1.f) * (float)Hin - 1.f) * 0.5f;
  int iy = (int)rintf(fy);
  const bool oky = (iy >= 0) && (iy < Hin);
  iy = min(max(iy, 0), Hin - 1);
  const float vx = -49.f + 2.f * (float)ox;
  const float gx = (vx + 51.2f) * (1.f / 102.4f) * 2.f - 1.f;
  const float fx = ((gx + 1.f) * (float)Win - 1.f) * 0.5f;
  int ix = (int)rintf(fx);
  const bool okx = (ix >= 0) && (ix < Win);
  ix = min(max(ix, 0), Win - 1);
  Out[idx] = (oky && okx) ? X[(long)c * (Hin * Win) + (long)iy * Win + ix] : 0.f;
}

__global__ __launch_bounds__(256) void pixel_shuffle_kernel(
    const float* __restrict__ In, float* __restrict__ Out,
    int C, int H, int W, int r)
{
  const int Ho = H * r, Wo = W * r;
  const int idx = blockIdx.x * blockDim.x + threadIdx.x;
  const int total = C * Ho * Wo;
  if (idx >= total) return;
  const int x = idx % Wo;
  const int t = idx / Wo;
  const int y = t % Ho;
  const int c = t / Ho;
  const int ci = c * r * r + (y % r) * r + (x % r);
  Out[idx] = In[(long)ci * (H * W) + (long)(y / r) * W + (x / r)];
}

// ---------------------------------------------------------------------------
// Host orchestration. ~108 MB workspace, bump-allocated; attention temporaries
// overlaid by conv buffers (phases strictly sequential on one stream).
// ---------------------------------------------------------------------------
extern "C" void kernel_launch(void* const* d_in, const int* in_sizes, int n_in,
                              void* d_out, int out_size, void* d_ws, size_t ws_size,
                              hipStream_t stream) {
  const float* x_in  = (const float*)d_in[0];
  const float* qkvw  = (const float*)d_in[1];
  const float* projw = (const float*)d_in[2];
  const float* projb = (const float*)d_in[3];
  const float* headw = (const float*)d_in[4];
  const float* headb = (const float*)d_in[5];
  const float* bodyw = (const float*)d_in[6];
  const float* bodyb = (const float*)d_in[7];
  const float* btailw= (const float*)d_in[8];
  const float* btailb= (const float*)d_in[9];
  const float* upw   = (const float*)d_in[10];
  const float* upb   = (const float*)d_in[11];
  const float* tailw = (const float*)d_in[12];
  const float* tailb = (const float*)d_in[13];
  const float* c1w   = (const float*)d_in[14];
  const float* bn1g  = (const float*)d_in[15];
  const float* bn1b  = (const float*)d_in[16];
  const float* bn1m  = (const float*)d_in[17];
  const float* bn1v  = (const float*)d_in[18];
  const float* c2w   = (const float*)d_in[19];
  const float* bn2g  = (const float*)d_in[20];
  const float* bn2b  = (const float*)d_in[21];
  const float* bn2m  = (const float*)d_in[22];
  const float* bn2v  = (const float*)d_in[23];
  const float* c3w   = (const float*)d_in[24];
  const float* c3b   = (const float*)d_in[25];

  const int C = 512, NTOK = 1024, OH = 50, OW = 50, M50 = OH * OW, M100 = 100 * 100;

  float* ws = (float*)d_ws;
  float* x_cur   = ws;
  const size_t o = (size_t)C * NTOK;
  float* qkv     = ws + o;
  float* scores  = qkv + (size_t)NTOK * 3 * C;
  float* attnout = scores + (size_t)8 * NTOK * NTOK;
  float* xs      = ws + o;
  float* buf_h   = xs     + (size_t)C * M50;
  float* buf_r   = buf_h  + (size_t)C * M50;
  float* buf_r2  = buf_r  + (size_t)C * M50;
  float* buf_t   = buf_r2 + (size_t)C * M50;
  float* buf_up  = buf_t  + (size_t)C * M50;
  float* buf_ps  = buf_up + (size_t)4 * C * M50;
  float* buf_y   = buf_ps + (size_t)C * M100;
  float* buf_z   = buf_y  + (size_t)C * M100;
  float* zp      = buf_z  + (size_t)C * M100;   // 256-float zero page

  hipMemsetAsync(zp, 0, 256 * sizeof(float), stream);
  hipMemcpyAsync(x_cur, x_in, (size_t)C * NTOK * sizeof(float),
                 hipMemcpyDeviceToDevice, stream);

  const float alpha_s = 0.044194173824159216f;  // 512^-0.5

  // ---- 8 self-attention blocks ----
  for (int i = 0; i < 8; ++i) {
    gemm_wmma_kernel<<<dim3(16, 12, 1), 256, 0, stream>>>(
        x_cur, 1, NTOK, 0,
        qkvw + (size_t)i * C * 3 * C, 3 * C, 1, 0,
        qkv, 3 * C, 1, 0,
        nullptr, nullptr, 0, 0, 0, zp,
        NTOK, 3 * C, C, 1.f, 0);
    gemm_wmma_kernel<<<dim3(16, 8, 8), 256, 0, stream>>>(
        qkv, 3 * C, 1, 64,
        qkv + C, 1, 3 * C, 64,
        scores, NTOK, 1, (long)NTOK * NTOK,
        nullptr, nullptr, 0, 0, 0, zp,
        NTOK, NTOK, 64, alpha_s, 0);
    softmax_rows_kernel<<<8 * NTOK, 256, 0, stream>>>(scores, NTOK);
    gemm_wmma_kernel<<<dim3(16, 1, 8), 256, 0, stream>>>(
        scores, NTOK, 1, (long)NTOK * NTOK,
        qkv + 2 * C, 3 * C, 1, 64,
        attnout, C, 1, 64,
        nullptr, nullptr, 0, 0, 0, zp,
        NTOK, 64, NTOK, 1.f, 0);
    gemm_wmma_kernel<<<dim3(16, 4, 1), 256, 0, stream>>>(
        attnout, C, 1, 0,
        projw + (size_t)i * C * C, C, 1, 0,
        x_cur, 1, NTOK, 0,
        projb + (size_t)i * C,
        x_cur, 1, NTOK, 0, zp,
        NTOK, C, C, 1.f, 0);
  }

  // ---- BEVGridTransform ----
  {
    const int total = C * M50;
    grid_sample_kernel<<<(total + 255) / 256, 256, 0, stream>>>(
        x_cur, xs, 32, 32, OH, OW, C);
  }

  const dim3 g50((M50 + 63) / 64, C / 128);
  const dim3 g100((M100 + 63) / 64, C / 128);

  conv3x3_wmma_kernel<<<g50, 256, 0, stream>>>(
      xs, headw, buf_h, headb, nullptr,
      nullptr, nullptr, nullptr, nullptr, zp, C, OH, OW, C, 0);

  float* rcur = buf_h;
  float* ralt = buf_r;
  for (int i = 0; i < 8; ++i) {
    conv3x3_wmma_kernel<<<g50, 256, 0, stream>>>(
        rcur, bodyw + (size_t)(2 * i) * C * C * 9, buf_t,
        bodyb + (size_t)(2 * i) * C, nullptr,
        nullptr, nullptr, nullptr, nullptr, zp, C, OH, OW, C, 1);
    conv3x3_wmma_kernel<<<g50, 256, 0, stream>>>(
        buf_t, bodyw + (size_t)(2 * i + 1) * C * C * 9, ralt,
        bodyb + (size_t)(2 * i + 1) * C, rcur,
        nullptr, nullptr, nullptr, nullptr, zp, C, OH, OW, C, 0);
    float* old = rcur; rcur = ralt; ralt = (old == buf_h) ? buf_r2 : old;
  }

  conv3x3_wmma_kernel<<<g50, 256, 0, stream>>>(
      rcur, btailw, buf_t, btailb, buf_h,
      nullptr, nullptr, nullptr, nullptr, zp, C, OH, OW, C, 0);

  conv3x3_wmma_kernel<<<dim3((M50 + 63) / 64, (4 * C) / 128), 256, 0, stream>>>(
      buf_t, upw, buf_up, upb, nullptr,
      nullptr, nullptr, nullptr, nullptr, zp, C, OH, OW, 4 * C, 0);
  {
    const int total = C * M100;
    pixel_shuffle_kernel<<<(total + 255) / 256, 256, 0, stream>>>(
        buf_up, buf_ps, C, OH, OW, 2);
  }

  conv3x3_wmma_kernel<<<g100, 256, 0, stream>>>(
      buf_ps, tailw, buf_y, tailb, nullptr,
      nullptr, nullptr, nullptr, nullptr, zp, C, 100, 100, C, 0);

  conv3x3_wmma_kernel<<<g100, 256, 0, stream>>>(
      buf_y, c1w, buf_z, nullptr, nullptr,
      bn1g, bn1b, bn1m, bn1v, zp, C, 100, 100, C, 1);
  conv3x3_wmma_kernel<<<g100, 256, 0, stream>>>(
      buf_z, c2w, buf_y, nullptr, nullptr,
      bn2g, bn2b, bn2m, bn2v, zp, C, 100, 100, C, 1);

  gemm_wmma_kernel<<<dim3((M100 + 63) / 64, 1, 1), 256, 0, stream>>>(
      buf_y, 1, M100, 0,
      c3w, 1, C, 0,
      (float*)d_out, 1, M100, 0,
      c3b, nullptr, 0, 0, 0, zp,
      M100, 6, C, 1.f, 2);
}
